// Transformer_26268019982785
// MI455X (gfx1250) — matrix operand order repaired and checked by reading, not compiled, not run on hardware
//
#include <hip/hip_runtime.h>
#include <hip/hip_bf16.h>
#include <math.h>

// ---------------- CDNA5 WMMA types ----------------
typedef __attribute__((ext_vector_type(16))) __bf16    v16bf;
typedef __attribute__((ext_vector_type(8)))  float     v8f;
typedef __attribute__((ext_vector_type(2)))  __bf16    v2bf;
typedef __attribute__((ext_vector_type(4)))  unsigned  uint4v;

union FragB { v16bf v; uint4v q[2]; unsigned u[8]; };

__device__ __forceinline__ unsigned short f2bf(float f) {
    unsigned u = __float_as_uint(f);
    u += 0x7FFFu + ((u >> 16) & 1u);          // round-to-nearest-even
    return (unsigned short)(u >> 16);
}
// pack two fp32 -> two bf16 in one dword (v_cvt_pk_bf16_f32 when available)
__device__ __forceinline__ unsigned f2bf2(float a, float bb) {
#if __has_builtin(__builtin_amdgcn_cvt_pk_bf16_f32)
    v2bf r = __builtin_amdgcn_cvt_pk_bf16_f32(a, bb);
    union { v2bf v; unsigned u; } cv; cv.v = r; return cv.u;
#else
    return (unsigned)f2bf(a) | ((unsigned)f2bf(bb) << 16);
#endif
}
__device__ __forceinline__ float bf2f(unsigned short h) {
    return __uint_as_float(((unsigned)h) << 16);
}

// A-fragment (16x32 bf16, MxK). Per ISA: lanes 0-15: VGPR0-3 -> K=0..7,
// VGPR4-7 -> K=16..23 ; lanes 16-31: +8.  Two contiguous 16B loads.
__device__ __forceinline__ v16bf load_a_bf(const unsigned short* p0, int stride,
                                           int row0, int lane) {
    FragB f;
    const unsigned short* p = p0 + (row0 + (lane & 15)) * stride + (lane >> 4) * 8;
    f.q[0] = *(const uint4v*)(p);
    f.q[1] = *(const uint4v*)(p + 16);
    return f.v;
}

// B-fragment (32x16 bf16, KxN), data stored as W[n][k] row-major:
// lanes 0-15 hold K=0..7,16..23, lanes 16-31 K=8..15,24..31.  Two contiguous 16B loads.
__device__ __forceinline__ v16bf load_b_bf(const unsigned short* p0, int stride,
                                           int col0, int lane) {
    FragB f;
    const unsigned short* p = p0 + (col0 + (lane & 15)) * stride + (lane >> 4) * 8;
    f.q[0] = *(const uint4v*)(p);
    f.q[1] = *(const uint4v*)(p + 16);
    return f.v;
}

__device__ __forceinline__ v8f wmma_bf16(v16bf a, v16bf b, v8f c) {
    return __builtin_amdgcn_wmma_f32_16x16x32_bf16(false, a, false, b,
                                                   (short)0, c, false, false);
}

// store one 16x16 f32 D-tile as bf16, row-major dst (paired conversion)
__device__ __forceinline__ void store_d_bf16(unsigned short* dst, int stride,
                                             int row0, int col, int lane, v8f c,
                                             float bias) {
    const int g = (lane >> 4) * 8;
    unsigned short* p = dst + (row0 + g) * stride + col;
#pragma unroll
    for (int r = 0; r < 8; r += 2) {
        unsigned pk = f2bf2(c[r] + bias, c[r + 1] + bias);
        p[(r + 0) * stride] = (unsigned short)(pk & 0xFFFFu);
        p[(r + 1) * stride] = (unsigned short)(pk >> 16);
    }
}

// ---------------- kernels ----------------
__global__ void k_cvt_bf16(const float* __restrict__ in,
                           unsigned short* __restrict__ out, int n) {
    int i = blockIdx.x * blockDim.x + threadIdx.x;
    if (i < n) out[i] = f2bf(in[i]);
}

// h0 = x @ proj_w^T + proj_b  : [4096 x 96] * [96 x 256] -> bf16 [4096 x 256]
__global__ __launch_bounds__(256) void k_proj(const unsigned short* __restrict__ xb,
                                              const unsigned short* __restrict__ wp,
                                              const float* __restrict__ bp,
                                              unsigned short* __restrict__ h0) {
    const int lane = threadIdx.x & 31, wave = threadIdx.x >> 5;
    const int tile = blockIdx.x * 8 + wave;   // 4096 tiles of 16x16
    const int tm = tile >> 4, tn = tile & 15;
    v8f c = {};
#pragma unroll
    for (int kk = 0; kk < 96; kk += 32) {
        v16bf a = load_a_bf(xb + kk, 96, tm * 16, lane);
        v16bf b = load_b_bf(wp + kk, 96, tn * 16, lane);
        c = wmma_bf16(a, b, c);
    }
    const int n = tn * 16 + (lane & 15);
    store_d_bf16(h0, 256, tm * 16, n, lane, c, bp[n]);
}

// residual + LayerNorm over one 64x256 tile in LDS; 4 lanes per row
__device__ __forceinline__ void ln_block(unsigned short* sY, const float* sAcc,
                                         const float* w, const float* bvec, int tid) {
    const int row = tid >> 2, q = tid & 3;
    const float* ar = sAcc + row * 256 + q * 64;
    unsigned short* yr = sY + row * 264 + q * 64;
    float vals[64];
    float s = 0.f, s2 = 0.f;
#pragma unroll
    for (int i = 0; i < 64; ++i) {
        float v = bf2f(yr[i]) + ar[i];
        vals[i] = v; s += v; s2 += v * v;
    }
    s  += __shfl_xor(s, 1);  s  += __shfl_xor(s, 2);   // 4-lane row group
    s2 += __shfl_xor(s2, 1); s2 += __shfl_xor(s2, 2);
    const float mean = s * (1.f / 256.f);
    const float var  = s2 * (1.f / 256.f) - mean * mean;
    const float inv  = rsqrtf(var + 1e-5f);
    const float* wq = w + q * 64;
    const float* bq = bvec + q * 64;
#pragma unroll
    for (int i = 0; i < 64; ++i)
        yr[i] = f2bf((vals[i] - mean) * inv * wq[i] + bq[i]);
}

// One workgroup = one (prefix p, batch b): full 2-layer encoder in LDS.
__global__ __launch_bounds__(256) void k_encoder(
    const unsigned short* __restrict__ h0,     // [4096][256] bf16
    const unsigned short* __restrict__ wip,    // [L][768][256] bf16
    const unsigned short* __restrict__ wop,    // [L][256][256] bf16
    const unsigned short* __restrict__ w1,     // [L][1024][256] bf16
    const unsigned short* __restrict__ w2,     // [L][256][1024] bf16
    const float* __restrict__ ipb, const float* __restrict__ opb,
    const float* __restrict__ ln1w, const float* __restrict__ ln1b,
    const float* __restrict__ b1,  const float* __restrict__ b2,
    const float* __restrict__ ln2w, const float* __restrict__ ln2b,
    float* __restrict__ out)
{
    extern __shared__ char smem[];
    unsigned short* sY = (unsigned short*)smem;   // 64 x 264 bf16 activations
    unsigned short* sV = sY + 64 * 264;           // 64 x 264 bf16 V^T per head
    unsigned short* sT = sV + 64 * 264;           // 64 x 776 bf16 scratch
    float* sAcc = (float*)(sT + 64 * 776);        // 64 x 256 fp32

    const int tid = threadIdx.x;
    const int lane = tid & 31, wave = tid >> 5;
    const int tmw  = wave & 3;                    // this wave's row tile
    const int half = wave >> 2;                   // this wave's N half
    const int b = blockIdx.x & 63;
    const int p = blockIdx.x >> 6;
    const int kv = p + 1;                         // valid keys 0..p

    // h0 -> sY (128-bit copies)
    for (int i = tid; i < 64 * 32; i += 256) {
        int t = i >> 5, c = i & 31;
        *(uint4v*)(sY + t * 264 + c * 8) =
            *(const uint4v*)(h0 + (b * 64 + t) * 256 + c * 8);
    }
    __syncthreads();

    for (int l = 0; l < 2; ++l) {
        const unsigned short* Wip = wip + l * (768 * 256);
        const unsigned short* Wop = wop + l * (256 * 256);
        const unsigned short* W1  = w1  + l * (1024 * 256);
        const unsigned short* W2  = w2  + l * (256 * 1024);
        const float* Bip = ipb + l * 768;
        const float* Bop = opb + l * 256;
        const float* B1  = b1  + l * 1024;
        const float* B2  = b2  + l * 256;

        // ---- QKV: cache A (y rows) once, stream 24 weight tiles ----
        {
            v16bf Ay[8];
#pragma unroll
            for (int s = 0; s < 8; ++s)
                Ay[s] = load_a_bf(sY + s * 32, 264, tmw * 16, lane);
            for (int tn = half * 24; tn < half * 24 + 24; ++tn) {
                v8f c = {};
#pragma unroll
                for (int s = 0; s < 8; ++s)
                    c = wmma_bf16(Ay[s], load_b_bf(Wip + s * 32, 256, tn * 16, lane), c);
                int n = tn * 16 + (lane & 15);
                float bias = Bip[n];
                int g = (lane >> 4) * 8;
                if (n < 512) {                         // Q | K
                    store_d_bf16(sT, 776, tmw * 16, n, lane, c, bias);
                } else {                               // V stored transposed
                    int hh = (n - 512) >> 6, dl = (n - 512) & 63;
#pragma unroll
                    for (int r = 0; r < 8; ++r)
                        sV[dl * 264 + hh * 64 + (tmw * 16 + r + g)] = f2bf(c[r] + bias);
                }
            }
        }
        __syncthreads();

        // ---- scores S = Q K^T * (1/8) -> sAcc ----
        for (int rg = wave; rg < 16; rg += 8) {        // (head, row tile)
            int hh = rg >> 2, tm = rg & 3;
            v16bf A0 = load_a_bf(sT + hh * 64,      776, tm * 16, lane);
            v16bf A1 = load_a_bf(sT + hh * 64 + 32, 776, tm * 16, lane);
#pragma unroll
            for (int tn = 0; tn < 4; ++tn) {
                v8f c = {};
                c = wmma_bf16(A0, load_b_bf(sT + 256 + hh * 64,      776, tn * 16, lane), c);
                c = wmma_bf16(A1, load_b_bf(sT + 256 + hh * 64 + 32, 776, tn * 16, lane), c);
                int n = tn * 16 + (lane & 15);
                int g = (lane >> 4) * 8;
#pragma unroll
                for (int r = 0; r < 8; ++r)
                    sAcc[(tm * 16 + r + g) * 256 + hh * 64 + n] = c[r] * 0.125f;
            }
        }
        __syncthreads();

        // ---- masked softmax; P bf16 overwrites Q region of sT ----
        {
            int t = tid >> 2, hh = tid & 3;
            float* s = sAcc + t * 256 + hh * 64;
            float mx = -1e30f;
            for (int k = 0; k < kv; ++k) mx = fmaxf(mx, s[k]);
            float sum = 0.f;
            for (int k = 0; k < kv; ++k) sum += __expf(s[k] - mx);
            float rs = 1.f / sum;
            unsigned short* prow = sT + t * 776 + hh * 64;
            for (int k = 0; k < 64; ++k)
                prow[k] = f2bf(k < kv ? __expf(s[k] - mx) * rs : 0.f);
        }
        __syncthreads();

        // ---- O = P @ V -> sT cols 256..511 (overwrites K region) ----
        for (int rg = wave; rg < 16; rg += 8) {
            int hh = rg >> 2, tm = rg & 3;
            v16bf A0 = load_a_bf(sT + hh * 64,      776, tm * 16, lane);
            v16bf A1 = load_a_bf(sT + hh * 64 + 32, 776, tm * 16, lane);
#pragma unroll
            for (int tn = 0; tn < 4; ++tn) {
                v8f c = {};
                c = wmma_bf16(A0, load_b_bf(sV + hh * 64,      264, tn * 16, lane), c);
                c = wmma_bf16(A1, load_b_bf(sV + hh * 64 + 32, 264, tn * 16, lane), c);
                int n = tn * 16 + (lane & 15);
                store_d_bf16(sT, 776, tm * 16, 256 + hh * 64 + n, lane, c, 0.f);
            }
        }
        __syncthreads();

        // ---- out-proj: sAcc = O @ Wop^T + Bop ----
        {
            v16bf Ao[8];
#pragma unroll
            for (int s = 0; s < 8; ++s)
                Ao[s] = load_a_bf(sT + 256 + s * 32, 776, tmw * 16, lane);
            for (int tn = half * 8; tn < half * 8 + 8; ++tn) {
                v8f c = {};
#pragma unroll
                for (int s = 0; s < 8; ++s)
                    c = wmma_bf16(Ao[s], load_b_bf(Wop + s * 32, 256, tn * 16, lane), c);
                int n = tn * 16 + (lane & 15);
                float bias = Bop[n];
                int g = (lane >> 4) * 8;
#pragma unroll
                for (int r = 0; r < 8; ++r)
                    sAcc[(tmw * 16 + r + g) * 256 + n] = c[r] + bias;
            }
        }
        __syncthreads();
        ln_block(sY, sAcc, ln1w + l * 256, ln1b + l * 256, tid);
        __syncthreads();

        // ---- FFN in two 512-wide chunks ----
        for (int ch = 0; ch < 2; ++ch) {
            {   // G = gelu(y W1_chunk^T + b1) -> sT cols 0..511
                v16bf Ay[8];
#pragma unroll
                for (int s = 0; s < 8; ++s)
                    Ay[s] = load_a_bf(sY + s * 32, 264, tmw * 16, lane);
                for (int tn = half * 16; tn < half * 16 + 16; ++tn) {
                    v8f c = {};
#pragma unroll
                    for (int s = 0; s < 8; ++s)
                        c = wmma_bf16(Ay[s],
                              load_b_bf(W1 + (ch * 512) * 256 + s * 32, 256, tn * 16, lane), c);
                    int n = tn * 16 + (lane & 15);
                    float bias = B1[ch * 512 + n];
                    int g = (lane >> 4) * 8;
#pragma unroll
                    for (int r = 0; r < 8; ++r) {
                        float xg = c[r] + bias;
                        c[r] = 0.5f * xg * (1.f + erff(xg * 0.70710678118f));
                    }
                    store_d_bf16(sT, 776, tmw * 16, n, lane, c, 0.f);
                }
            }
            __syncthreads();
            {   // sAcc (+)= G @ W2_chunk^T   (b2 on first chunk)
                v16bf Ag[16];
#pragma unroll
                for (int s = 0; s < 16; ++s)
                    Ag[s] = load_a_bf(sT + s * 32, 776, tmw * 16, lane);
                for (int tn = half * 8; tn < half * 8 + 8; ++tn) {
                    v8f c = {};
#pragma unroll
                    for (int s = 0; s < 16; ++s)
                        c = wmma_bf16(Ag[s],
                              load_b_bf(W2 + ch * 512 + s * 32, 1024, tn * 16, lane), c);
                    int n = tn * 16 + (lane & 15);
                    int g = (lane >> 4) * 8;
#pragma unroll
                    for (int r = 0; r < 8; ++r) {
                        int idx = (tmw * 16 + r + g) * 256 + n;
                        float prev = (ch == 0) ? B2[n] : sAcc[idx];
                        sAcc[idx] = prev + c[r];
                    }
                }
            }
            __syncthreads();
        }
        ln_block(sY, sAcc, ln2w + l * 256, ln2b + l * 256, tid);
        __syncthreads();
    }

    // ---- prefix-mean pooling: out[b, p, h] ----
    {
        float s = 0.f;
        for (int t = 0; t < kv; ++t) s += bf2f(sY[t * 264 + tid]);
        out[(b * 64 + p) * 256 + tid] = s / (float)kv;
    }
}

// ---------------- launch ----------------
extern "C" void kernel_launch(void* const* d_in, const int* in_sizes, int n_in,
                              void* d_out, int out_size, void* d_ws, size_t ws_size,
                              hipStream_t stream) {
    const float* x      = (const float*)d_in[0];
    const float* proj_w = (const float*)d_in[1];
    const float* proj_b = (const float*)d_in[2];
    const float* ipw    = (const float*)d_in[3];
    const float* ipb    = (const float*)d_in[4];
    const float* opw    = (const float*)d_in[5];
    const float* opb    = (const float*)d_in[6];
    const float* ln1w   = (const float*)d_in[7];
    const float* ln1b   = (const float*)d_in[8];
    const float* w1     = (const float*)d_in[9];
    const float* b1     = (const float*)d_in[10];
    const float* w2     = (const float*)d_in[11];
    const float* b2     = (const float*)d_in[12];
    const float* ln2w   = (const float*)d_in[13];
    const float* ln2b   = (const float*)d_in[14];
    float* out = (float*)d_out;

    unsigned short* ws = (unsigned short*)d_ws;
    unsigned short* wt_proj = ws;            size_t o = 256 * 96;
    unsigned short* wt_ipw  = ws + o;        o += 2 * 768 * 256;
    unsigned short* wt_opw  = ws + o;        o += 2 * 256 * 256;
    unsigned short* wt_w1   = ws + o;        o += 2 * 1024 * 256;
    unsigned short* wt_w2   = ws + o;        o += 2 * 256 * 1024;
    unsigned short* xb      = ws + o;        o += 4096 * 96;   // x in bf16
    unsigned short* h0      = ws + o;        // 4096 x 256 bf16

    k_cvt_bf16<<<(256 * 96 + 255) / 256, 256, 0, stream>>>(proj_w, wt_proj, 256 * 96);
    k_cvt_bf16<<<(2 * 768 * 256 + 255) / 256, 256, 0, stream>>>(ipw, wt_ipw, 2 * 768 * 256);
    k_cvt_bf16<<<(2 * 256 * 256 + 255) / 256, 256, 0, stream>>>(opw, wt_opw, 2 * 256 * 256);
    k_cvt_bf16<<<(2 * 1024 * 256 + 255) / 256, 256, 0, stream>>>(w1, wt_w1, 2 * 1024 * 256);
    k_cvt_bf16<<<(2 * 256 * 1024 + 255) / 256, 256, 0, stream>>>(w2, wt_w2, 2 * 256 * 1024);
    k_cvt_bf16<<<(4096 * 96 + 255) / 256, 256, 0, stream>>>(x, xb, 4096 * 96);

    k_proj<<<512, 256, 0, stream>>>(xb, wt_proj, proj_b, h0);

    const size_t shmem = (size_t)(64 * 264 + 64 * 264 + 64 * 776) * 2
                       + (size_t)64 * 256 * 4;   // 232448 B < 320 KB
    (void)hipFuncSetAttribute((const void*)k_encoder,
                              hipFuncAttributeMaxDynamicSharedMemorySize, (int)shmem);
    k_encoder<<<4096, 256, shmem, stream>>>(h0, wt_ipw, wt_opw, wt_w1, wt_w2,
                                            ipb, opb, ln1w, ln1b, b1, b2,
                                            ln2w, ln2b, out);
}
